// GROVERMoE_62053687493030
// MI455X (gfx1250) — compile-verified, matrix operand order checked by reading, not run on hardware
//
#include <hip/hip_runtime.h>
#include <hip/hip_bf16.h>
#include <stdint.h>

#define DIM  1024
#define NEXP 8
#define NTOK 4096
#define HID  4096
#define THRESH 0.3f

typedef __attribute__((ext_vector_type(16))) __bf16 v16bf;
typedef __attribute__((ext_vector_type(8)))  float  v8f;

// Block tile: 256(M) x 128(N), K-step 32. 8 waves arranged 4(M) x 2(N),
// each wave computes a 64x64 tile = 4x4 WMMA fragments (16 wmma / K-step).
#define BM 256
#define BN 128
#define BK 32
#define LDS_STRIDE 34   // 32 + 2 pad (even); 17-dword pitch, conflict-friendly

// ---------------------------------------------------------------- utilities
__device__ __forceinline__ uint16_t f2bf(float x) {
    uint32_t u = __builtin_bit_cast(uint32_t, x);
    uint32_t r = u + 0x7FFFu + ((u >> 16) & 1u);   // round-to-nearest-even
    return (uint16_t)(r >> 16);
}

union FragCvt { uint32_t u[8]; v16bf v; };

// Assemble a 16-bit WMMA operand fragment from an LDS tile stored [i][k]
// (i = M for A, N for B), row stride in uint16 units (must be even).
// CDNA5 ISA 7.12.2: lane L (i = L%16, h = L/16), VGPR v holds
// K = (v<4?0:16) + h*8 + (v%4)*2 and K+1.
__device__ __forceinline__ v16bf load_frag(const uint16_t* lds, int base_i, int stride) {
    const int lane = threadIdx.x & 31;
    const int i = lane & 15;
    const int h = lane >> 4;
    const uint16_t* rowp = lds + (size_t)(base_i + i) * stride;
    FragCvt f;
#pragma unroll
    for (int v = 0; v < 8; ++v) {
        int kb = ((v & 3) << 1) + (h << 3) + ((v >> 2) << 4);
        f.u[v] = *reinterpret_cast<const uint32_t*>(rowp + kb);
    }
    return f.v;
}

// ---------------------------------------------------------------- zero output
__global__ __launch_bounds__(256) void moe_zero_kernel(float* __restrict__ out, int n) {
    int idx = blockIdx.x * 256 + threadIdx.x;
    if (idx < n) out[idx] = 0.0f;
}

// ---------------------------------------------------------------- gate
__global__ __launch_bounds__(256) void moe_gate_kernel(
    const float* __restrict__ x, const float* __restrict__ wg,
    const float* __restrict__ bg, float* __restrict__ gate_out,
    float* __restrict__ cws) {
    const int n    = blockIdx.x;
    const int wave = threadIdx.x >> 5;
    const int lane = threadIdx.x & 31;

    float p = 0.0f;
    const float* xr = x + (size_t)n * DIM;
    for (int k = lane; k < DIM; k += 32)
        p += xr[k] * wg[k * NEXP + wave];
#pragma unroll
    for (int o = 16; o > 0; o >>= 1)
        p += __shfl_xor(p, o, 32);

    __shared__ float sc[NEXP];
    if (lane == 0) sc[wave] = p + bg[wave];
    __syncthreads();

    if (threadIdx.x == 0) {
        float s[NEXP];
        float m = sc[0];
#pragma unroll
        for (int e = 1; e < NEXP; ++e) m = fmaxf(m, sc[e]);
        float denom = 0.0f;
#pragma unroll
        for (int e = 0; e < NEXP; ++e) { s[e] = __expf(sc[e] - m); denom += s[e]; }
        float inv = 1.0f / denom;
        float msum = 0.0f; int cnt = 0; int top1 = 0; float best = -1.0f;
#pragma unroll
        for (int e = 0; e < NEXP; ++e) {
            s[e] *= inv;
            if (s[e] > best) { best = s[e]; top1 = e; }
            if (s[e] >= THRESH) { msum += s[e]; ++cnt; }
        }
        float rn = 1.0f / (msum + 1e-6f);
#pragma unroll
        for (int e = 0; e < NEXP; ++e) {
            gate_out[(size_t)n * NEXP + e] = s[e];
            float c;
            if (cnt > 0) c = (s[e] >= THRESH) ? s[e] * rn : 0.0f;
            else         c = (e == top1) ? 1.0f : 0.0f;
            cws[(size_t)n * NEXP + e] = c;
        }
    }
}

// ---------------------------------------------------------------- GEMM1+GELU
// h[e] = gelu(expert_inputs[e] @ w1[e] + b1[e]), bf16 out.
__global__ __launch_bounds__(256) void moe_gemm1_kernel(
    const float* __restrict__ xin, const float* __restrict__ w1,
    const float* __restrict__ b1, uint16_t* __restrict__ hout) {
    const int e    = blockIdx.z;
    const int row0 = blockIdx.y * BM;
    const int n0   = blockIdx.x * BN;
    const int tid  = threadIdx.x;
    const int wv = tid >> 5, wm = wv >> 1, wn = wv & 1;

    const float* A  = xin + (size_t)e * NTOK * DIM;
    const float* B  = w1  + (size_t)e * DIM * HID;
    const float* be = b1  + (size_t)e * HID;
    uint16_t* H     = hout + (size_t)e * NTOK * HID;

    __shared__ uint16_t sA[BM * LDS_STRIDE];
    __shared__ uint16_t sB[BN * LDS_STRIDE];

    v8f acc[4][4];
#pragma unroll
    for (int i = 0; i < 4; ++i)
#pragma unroll
        for (int j = 0; j < 4; ++j)
#pragma unroll
            for (int r = 0; r < 8; ++r) acc[i][j][r] = 0.0f;

    for (int kk = 0; kk < DIM / BK; ++kk) {
        const int k0 = kk * BK;
        // A tile: 256 rows x 32 k (fp32 -> bf16), row-major in LDS
#pragma unroll
        for (int it = 0; it < 8; ++it) {
            int s = it * 256 + tid;           // 2048 float4 slots
            int row = s >> 3, c4 = s & 7;
            float4 v = *reinterpret_cast<const float4*>(
                A + (size_t)(row0 + row) * DIM + k0 + c4 * 4);
            uint32_t* dst = reinterpret_cast<uint32_t*>(&sA[row * LDS_STRIDE + c4 * 4]);
            dst[0] = (uint32_t)f2bf(v.x) | ((uint32_t)f2bf(v.y) << 16);
            dst[1] = (uint32_t)f2bf(v.z) | ((uint32_t)f2bf(v.w) << 16);
        }
        // B tile: 32 k x 128 n (fp32 -> bf16), stored transposed [n][k]
#pragma unroll
        for (int it = 0; it < 4; ++it) {
            int s = it * 256 + tid;           // 1024 float4 slots
            int kr = s >> 5, nc4 = s & 31;
            float4 v = *reinterpret_cast<const float4*>(
                B + (size_t)(k0 + kr) * HID + n0 + nc4 * 4);
            sB[(nc4 * 4 + 0) * LDS_STRIDE + kr] = f2bf(v.x);
            sB[(nc4 * 4 + 1) * LDS_STRIDE + kr] = f2bf(v.y);
            sB[(nc4 * 4 + 2) * LDS_STRIDE + kr] = f2bf(v.z);
            sB[(nc4 * 4 + 3) * LDS_STRIDE + kr] = f2bf(v.w);
        }
        __syncthreads();

        v16bf af[4], bfr[4];
#pragma unroll
        for (int tm = 0; tm < 4; ++tm) af[tm]  = load_frag(sA, wm * 64 + tm * 16, LDS_STRIDE);
#pragma unroll
        for (int tn = 0; tn < 4; ++tn) bfr[tn] = load_frag(sB, wn * 64 + tn * 16, LDS_STRIDE);
#pragma unroll
        for (int tm = 0; tm < 4; ++tm)
#pragma unroll
            for (int tn = 0; tn < 4; ++tn)
                acc[tm][tn] = __builtin_amdgcn_wmma_f32_16x16x32_bf16(
                    false, af[tm], false, bfr[tn], (short)0, acc[tm][tn], false, false);
        __syncthreads();
    }

    // epilogue: + b1, exact GELU, store bf16
    const int lane = threadIdx.x & 31;
    const int ln = lane & 15, lh = lane >> 4;
#pragma unroll
    for (int tm = 0; tm < 4; ++tm)
#pragma unroll
        for (int tn = 0; tn < 4; ++tn) {
            int col = n0 + wn * 64 + tn * 16 + ln;
            float bias = be[col];
#pragma unroll
            for (int r = 0; r < 8; ++r) {
                int row = row0 + wm * 64 + tm * 16 + lh * 8 + r;
                float xv = acc[tm][tn][r] + bias;
                float g = 0.5f * xv * (1.0f + erff(xv * 0.70710678118f));
                H[(size_t)row * HID + col] = f2bf(g);
            }
        }
}

// ---------------------------------------------------------------- GEMM2+combine
// out += c[n][e] * (h[e] @ w2[e] + b2[e]).  A-tile staged with CDNA5 async
// global->LDS copies (ASYNCcnt), B-tile converted fp32->bf16 through VGPRs.
__global__ __launch_bounds__(256) void moe_gemm2_kernel(
    const uint16_t* __restrict__ hin, const float* __restrict__ w2,
    const float* __restrict__ b2, const float* __restrict__ cws,
    float* __restrict__ out) {
    const int e    = blockIdx.z;
    const int row0 = blockIdx.y * BM;
    const int n0   = blockIdx.x * BN;
    const int tid  = threadIdx.x;
    const int wv = tid >> 5, wm = wv >> 1, wn = wv & 1;

    const uint16_t* A = hin + (size_t)e * NTOK * HID;
    const float* B    = w2  + (size_t)e * HID * DIM;
    const float* be   = b2  + (size_t)e * DIM;

    __shared__ uint16_t sA[BM * LDS_STRIDE];
    __shared__ uint16_t sB[BN * LDS_STRIDE];

    v8f acc[4][4];
#pragma unroll
    for (int i = 0; i < 4; ++i)
#pragma unroll
        for (int j = 0; j < 4; ++j)
#pragma unroll
            for (int r = 0; r < 8; ++r) acc[i][j][r] = 0.0f;

    for (int kk = 0; kk < HID / BK; ++kk) {
        const int k0 = kk * BK;
        // A tile: 256 x 32 bf16 — async global->LDS B128 copies (16B per lane-op)
#pragma unroll
        for (int it = 0; it < 4; ++it) {
            int s = it * 256 + tid;           // 1024 b128 slots (8 bf16 each)
            int row = s >> 2, c8 = s & 3;
            uint64_t gaddr = (uint64_t)(uintptr_t)(
                A + (size_t)(row0 + row) * HID + k0 + c8 * 8);
            uint32_t laddr = (uint32_t)(uintptr_t)(&sA[row * LDS_STRIDE + c8 * 8]);
            asm volatile("global_load_async_to_lds_b128 %0, %1, off"
                         :: "v"(laddr), "v"(gaddr) : "memory");
        }
        // B tile: 32 k x 128 n fp32 -> bf16, transposed [n][k]
#pragma unroll
        for (int it = 0; it < 4; ++it) {
            int s = it * 256 + tid;
            int kr = s >> 5, nc4 = s & 31;
            float4 v = *reinterpret_cast<const float4*>(
                B + (size_t)(k0 + kr) * DIM + n0 + nc4 * 4);
            sB[(nc4 * 4 + 0) * LDS_STRIDE + kr] = f2bf(v.x);
            sB[(nc4 * 4 + 1) * LDS_STRIDE + kr] = f2bf(v.y);
            sB[(nc4 * 4 + 2) * LDS_STRIDE + kr] = f2bf(v.z);
            sB[(nc4 * 4 + 3) * LDS_STRIDE + kr] = f2bf(v.w);
        }
        // drain this wave's async copies before the workgroup barrier
        asm volatile("s_wait_asynccnt 0x0" ::: "memory");
        __syncthreads();

        v16bf af[4], bfr[4];
#pragma unroll
        for (int tm = 0; tm < 4; ++tm) af[tm]  = load_frag(sA, wm * 64 + tm * 16, LDS_STRIDE);
#pragma unroll
        for (int tn = 0; tn < 4; ++tn) bfr[tn] = load_frag(sB, wn * 64 + tn * 16, LDS_STRIDE);
#pragma unroll
        for (int tm = 0; tm < 4; ++tm)
#pragma unroll
            for (int tn = 0; tn < 4; ++tn)
                acc[tm][tn] = __builtin_amdgcn_wmma_f32_16x16x32_bf16(
                    false, af[tm], false, bfr[tn], (short)0, acc[tm][tn], false, false);
        __syncthreads();
    }

    const int lane = threadIdx.x & 31;
    const int ln = lane & 15, lh = lane >> 4;
#pragma unroll
    for (int tm = 0; tm < 4; ++tm)
#pragma unroll
        for (int tn = 0; tn < 4; ++tn) {
            int col = n0 + wn * 64 + tn * 16 + ln;
            float bias = be[col];
#pragma unroll
            for (int r = 0; r < 8; ++r) {
                int row = row0 + wm * 64 + tm * 16 + lh * 8 + r;
                float cw = cws[(size_t)row * NEXP + e];
                float v = cw * (acc[tm][tn][r] + bias);
                unsafeAtomicAdd(&out[(size_t)row * DIM + col], v);
            }
        }
}

// ---------------------------------------------------------------- launch
extern "C" void kernel_launch(void* const* d_in, const int* in_sizes, int n_in,
                              void* d_out, int out_size, void* d_ws, size_t ws_size,
                              hipStream_t stream) {
    const float* x   = (const float*)d_in[0];   // [N, DIM]
    const float* xin = (const float*)d_in[1];   // [E, N, DIM]
    const float* wg  = (const float*)d_in[2];   // [DIM, E]
    const float* bg  = (const float*)d_in[3];   // [E]
    const float* w1  = (const float*)d_in[4];   // [E, DIM, HID]
    const float* b1  = (const float*)d_in[5];   // [E, HID]
    const float* w2  = (const float*)d_in[6];   // [E, HID, DIM]
    const float* b2  = (const float*)d_in[7];   // [E, DIM]

    float* fused = (float*)d_out;                       // [N, DIM]
    float* gate  = fused + (size_t)NTOK * DIM;          // [N, E]

    // workspace: h (bf16, E*N*HID) then combine weights c (f32, N*E)
    uint16_t* h = (uint16_t*)d_ws;
    float* cws  = (float*)((char*)d_ws + (size_t)NEXP * NTOK * HID * sizeof(uint16_t));

    moe_zero_kernel<<<(NTOK * DIM + 255) / 256, 256, 0, stream>>>(fused, NTOK * DIM);
    moe_gate_kernel<<<NTOK, 256, 0, stream>>>(x, wg, bg, gate, cws);

    dim3 g1(HID / BN, NTOK / BM, NEXP);
    moe_gemm1_kernel<<<g1, 256, 0, stream>>>(xin, w1, b1, h);

    dim3 g2(DIM / BN, NTOK / BM, NEXP);
    moe_gemm2_kernel<<<g2, 256, 0, stream>>>(h, w2, b2, cws, fused);
}
